// DGCNN_Mod_31877247271068
// MI455X (gfx1250) — matrix operand order repaired and checked by reading, not compiled, not run on hardware
//
#include <hip/hip_runtime.h>
#include <hip/hip_bf16.h>
#include <stdint.h>

// ---------------------------------------------------------------------------
// DGCNN stem for MI455X (gfx1250): wave32 + v_wmma_f32_16x16x32_bf16 GEMMs.
// Pipeline:
//   knn -> edgeconv1(VALU, Cin=6) -> [stats -> bnparam -> act+maxpool] ->
//   wmma GEMM (64->64) -> ... -> wmma GEMM (512->512 over cat) -> transpose out
// Activations stored bf16 row-major [row][channel] so WMMA fragments are
// contiguous 16B loads per lane. Weight tiles staged in LDS (ds path).
// ---------------------------------------------------------------------------

typedef __bf16 bf16_t;
typedef __attribute__((ext_vector_type(8)))  __bf16 v8bf;
typedef __attribute__((ext_vector_type(16))) __bf16 v16bf;
typedef __attribute__((ext_vector_type(8)))  float  v8f;

#define KNN 20

// ------------------------------- utility -----------------------------------

__global__ void zero_f32_kernel(float* __restrict__ p, int n) {
    int i = blockIdx.x * blockDim.x + threadIdx.x;
    if (i < n) p[i] = 0.0f;
}

__global__ void cvt_f32_to_bf16_kernel(const float* __restrict__ src,
                                       bf16_t* __restrict__ dst, int n) {
    int i = blockIdx.x * blockDim.x + threadIdx.x;
    if (i < n) dst[i] = (bf16_t)src[i];
}

// ------------------------------- kNN ---------------------------------------
// One block = 256 points of one batch; the batch's 3x2048 coords live in LDS.
__global__ __launch_bounds__(256) void knn_kernel(const float* __restrict__ x,
                                                  int* __restrict__ idx,
                                                  int B, int N) {
    __shared__ float xs0[2048], xs1[2048], xs2[2048];
    const int blocksPerBatch = N / 256;
    const int b = blockIdx.x / blocksPerBatch;
    const int nbase = (blockIdx.x % blocksPerBatch) * 256;
    const float* xb = x + (size_t)b * 3 * N;
    for (int m = threadIdx.x; m < N; m += 256) {
        xs0[m] = xb[m];
        xs1[m] = xb[N + m];
        xs2[m] = xb[2 * N + m];
    }
    __syncthreads();

    const int n = nbase + threadIdx.x;
    const float cx = xs0[n], cy = xs1[n], cz = xs2[n];

    float vals[KNN];
    int   ids[KNN];
    for (int j = 0; j < KNN; ++j) { vals[j] = -1e30f; ids[j] = 0; }

    for (int m = 0; m < N; ++m) {
        float dx = xs0[m] - cx, dy = xs1[m] - cy, dz = xs2[m] - cz;
        float d = -(dx * dx + dy * dy + dz * dz);   // neg squared distance
        if (d > vals[KNN - 1]) {
            int p = KNN - 1;
            while (p > 0 && vals[p - 1] < d) {
                vals[p] = vals[p - 1]; ids[p] = ids[p - 1]; --p;
            }
            vals[p] = d; ids[p] = m;
        }
    }
    int* op = idx + ((size_t)b * N + n) * KNN;
    for (int j = 0; j < KNN; ++j) op[j] = ids[j];
}

// --------------------------- layer 1 (Cin=6) --------------------------------
// Builds the edge feature [nbr-ctr, ctr] on the fly and applies W1 (64x6).
__global__ __launch_bounds__(128) void edgeconv1_kernel(
    const float* __restrict__ x, const int* __restrict__ idx,
    const float* __restrict__ W1, bf16_t* __restrict__ y1, int B, int N) {
    __shared__ float w[64 * 6];
    for (int i = threadIdx.x; i < 64 * 6; i += blockDim.x) w[i] = W1[i];
    __syncthreads();

    const long E = (long)B * N * KNN;
    long e = (long)blockIdx.x * blockDim.x + threadIdx.x;
    if (e >= E) return;
    long s = e / KNN;
    int b = (int)(s / N), n = (int)(s % N);
    int j = idx[e];
    const float* xb = x + (size_t)b * 3 * N;
    float c0 = xb[n],     c1 = xb[N + n],     c2 = xb[2 * N + n];
    float f0 = xb[j] - c0, f1 = xb[N + j] - c1, f2 = xb[2 * N + j] - c2;

    bf16_t* yo = y1 + e * 64;
    for (int o8 = 0; o8 < 64; o8 += 8) {
        union { float4 f; bf16_t h[8]; } st;
        #pragma unroll
        for (int r = 0; r < 8; ++r) {
            const float* wo = w + (o8 + r) * 6;
            st.h[r] = (bf16_t)(wo[0] * f0 + wo[1] * f1 + wo[2] * f2 +
                               wo[3] * c0 + wo[4] * c1 + wo[5] * c2);
        }
        *(float4*)(yo + o8) = st.f;
    }
}

// --------------------------- WMMA GEMM --------------------------------------
// Y[rows][Cout] = W[Cout][Cin] * X[rows][Cin]^T, bf16 in, raw bf16 out.
// Block = 128 threads = 4 waves; each wave owns one 16-row tile and 4 M-tiles
// (64 output channels) so each B-fragment feeds 4 wmma ops. W tile in LDS.
__global__ __launch_bounds__(128) void wmma_gemm_bf16_kernel(
    const bf16_t* __restrict__ W, const bf16_t* __restrict__ X,
    bf16_t* __restrict__ Y, int Cin, int Cout, int rows) {
    extern __shared__ __align__(16) bf16_t ldsW[];   // [64][Cin]

    const int tid   = threadIdx.x;
    const int oBase = blockIdx.y * 64;

    // Cooperative stage of 64 weight rows into LDS (contiguous copy).
    {
        const v8bf* wsrc = (const v8bf*)(W + (size_t)oBase * Cin);
        v8bf* wdst = (v8bf*)ldsW;
        const int totalVec = (64 * Cin) / 8;
        for (int i = tid; i < totalVec; i += 128) wdst[i] = wsrc[i];
    }
    __syncthreads();

    const int wave = tid >> 5;
    const int lane = tid & 31;
    const int half = lane >> 4;
    const int l16  = lane & 15;

    const long e0 = ((long)blockIdx.x * 4 + wave) * 16;   // 16-row tile

    v8f acc[4];
    #pragma unroll
    for (int mt = 0; mt < 4; ++mt) acc[mt] = (v8f){};

    const bf16_t* xrow = X + (size_t)(e0 + l16) * Cin;

    for (int kk = 0; kk < Cin; kk += 32) {
        // B fragment: column = e0+l16, K = half*16 .. half*16+15 (contiguous).
        union { v16bf v; v8bf h[2]; } bfr;
        const v8bf* bp = (const v8bf*)(xrow + kk + half * 16);
        bfr.h[0] = bp[0];
        bfr.h[1] = bp[1];
        if (kk + 32 < Cin)
            __builtin_prefetch((const void*)(bp + 4), 0, 1);   // next K chunk

        #pragma unroll
        for (int mt = 0; mt < 4; ++mt) {
            // A fragment: row = mt*16+l16, K = half*16 .. half*16+15 (contiguous).
            union { v16bf v; v8bf h[2]; } afr;
            const bf16_t* ab = ldsW + (size_t)(mt * 16 + l16) * Cin + kk + half * 16;
            afr.h[0] = *(const v8bf*)(ab);
            afr.h[1] = *(const v8bf*)(ab + 8);
            acc[mt] = __builtin_amdgcn_wmma_f32_16x16x32_bf16(
                false, afr.v, false, bfr.v, (short)0, acc[mt], false, false);
        }
    }

    // D tile: lane holds 8 contiguous channels (oBase+mt*16+half*8 .. +7) of
    // row e0+l16 -> one packed 16B bf16 store per M-tile.
    bf16_t* yrow = Y + (size_t)(e0 + l16) * Cout + oBase;
    #pragma unroll
    for (int mt = 0; mt < 4; ++mt) {
        union { float4 f; bf16_t h[8]; } st;
        #pragma unroll
        for (int r = 0; r < 8; ++r) st.h[r] = (bf16_t)acc[mt][r];
        *(float4*)(yrow + mt * 16 + half * 8) = st.f;
    }
}

// --------------------------- BN statistics ----------------------------------
// Per-channel sum/sumsq over a [rows][C] bf16 buffer (C in {64,128,256,512}).
// Element stride 512 keeps each thread on <=2 fixed channels.
__global__ __launch_bounds__(256) void stats_kernel(
    const bf16_t* __restrict__ y, long total, int C, float* __restrict__ stats) {
    long chunks = total / 512;                       // total is a multiple of 512
    long perBlk = (chunks + gridDim.x - 1) / gridDim.x;
    long start  = (long)blockIdx.x * perBlk * 512;
    if (start >= total) return;
    long end = start + perBlk * 512;
    if (end > total) end = total;

    const int t  = threadIdx.x;
    const int cA = (int)((start + t) % C);
    const int cB = (int)((start + t + 256) % C);

    float sA = 0.f, qA = 0.f, sB = 0.f, qB = 0.f;
    for (long i = start + t; i < end; i += 512) {
        float v = (float)y[i];
        sA += v; qA += v * v;
        long i2 = i + 256;
        if (i2 < end) {
            float v2 = (float)y[i2];
            sB += v2; qB += v2 * v2;
        }
    }
    atomicAdd(&stats[cA], sA);
    atomicAdd(&stats[C + cA], qA);
    atomicAdd(&stats[cB], sB);
    atomicAdd(&stats[C + cB], qB);
}

// Fold BN into per-channel scale/shift: s = g*rsqrt(var+eps), t = b - m*s.
__global__ void bnparam_kernel(const float* __restrict__ stats,
                               const float* __restrict__ g,
                               const float* __restrict__ b,
                               float cnt, int C, float* __restrict__ scbi) {
    int c = blockIdx.x * blockDim.x + threadIdx.x;
    if (c >= C) return;
    float m = stats[c] / cnt;
    float v = stats[C + c] / cnt - m * m;
    float s = g[c] * rsqrtf(v + 1e-5f);
    scbi[c] = s;
    scbi[C + c] = b[c] - m * s;
}

// Activate raw y in place (BN+ReLU) and max-pool over k into cat[s][catOff+c].
__global__ __launch_bounds__(256) void act_max_kernel(
    bf16_t* __restrict__ y, const float* __restrict__ scbi,
    bf16_t* __restrict__ cat, int C, int catOff, int S) {
    long t = (long)blockIdx.x * blockDim.x + threadIdx.x;
    if (t >= (long)S * C) return;
    int  c = (int)(t % C);
    long s = t / C;
    const float sc = scbi[c], bi = scbi[C + c];
    bf16_t* base = y + (size_t)s * KNN * C + c;
    float mx = 0.0f;                                 // ReLU outputs are >= 0
    for (int k = 0; k < KNN; ++k) {
        float v = fmaxf((float)base[k * C] * sc + bi, 0.0f);
        base[k * C] = (bf16_t)v;
        if (v > mx) mx = v;
    }
    cat[s * 512 + catOff + c] = (bf16_t)mx;
}

// Final BN+ReLU on y5 + LDS-tiled transpose [S][512] -> out [B][512][N] f32.
__global__ __launch_bounds__(256) void final_kernel(
    const bf16_t* __restrict__ y5, const float* __restrict__ scbi,
    float* __restrict__ out, int N) {
    __shared__ float tile[64][65];
    const int b = blockIdx.z;
    const int oBase = blockIdx.y * 64;
    const int nBase = blockIdx.x * 64;

    for (int l = threadIdx.x; l < 64 * 64; l += 256) {
        int i = l / 64, j = l % 64;                  // i: point, j: channel
        long s = (long)b * N + nBase + i;
        int  c = oBase + j;
        float v = (float)y5[s * 512 + c];
        tile[j][i] = fmaxf(v * scbi[c] + scbi[512 + c], 0.0f);
    }
    __syncthreads();
    for (int l = threadIdx.x; l < 64 * 64; l += 256) {
        int j = l / 64, i = l % 64;                  // i fastest -> coalesced
        out[((long)b * 512 + oBase + j) * N + nBase + i] = tile[j][i];
    }
}

// ------------------------------- driver -------------------------------------

extern "C" void kernel_launch(void* const* d_in, const int* in_sizes, int n_in,
                              void* d_out, int out_size, void* d_ws, size_t ws_size,
                              hipStream_t stream) {
    (void)in_sizes; (void)n_in; (void)out_size; (void)ws_size;

    const float* x  = (const float*)d_in[0];
    const float* Wf[5]; const float* gf[5]; const float* bf[5];
    for (int i = 0; i < 5; ++i) {
        Wf[i] = (const float*)d_in[1 + 3 * i];
        gf[i] = (const float*)d_in[2 + 3 * i];
        bf[i] = (const float*)d_in[3 + 3 * i];
    }
    float* out = (float*)d_out;

    const int B = 8, N = 2048, S = B * N;
    const long E = (long)S * KNN;                    // 327680 edges

    // --- carve workspace ---
    char* p = (char*)d_ws;
    auto alloc = [&](size_t bytes) -> void* {
        void* r = (void*)p;
        p += (bytes + 255) & ~(size_t)255;
        return r;
    };
    int*    idx  = (int*)alloc(E * sizeof(int));
    bf16_t* Wb2  = (bf16_t*)alloc((size_t)64 * 64 * 2);
    bf16_t* Wb3  = (bf16_t*)alloc((size_t)128 * 64 * 2);
    bf16_t* Wb4  = (bf16_t*)alloc((size_t)256 * 128 * 2);
    bf16_t* Wb5  = (bf16_t*)alloc((size_t)512 * 512 * 2);
    bf16_t* a1   = (bf16_t*)alloc((size_t)E * 64 * 2);
    bf16_t* a2   = (bf16_t*)alloc((size_t)E * 64 * 2);
    bf16_t* a3   = (bf16_t*)alloc((size_t)E * 128 * 2);
    bf16_t* a4   = (bf16_t*)alloc((size_t)E * 256 * 2);
    bf16_t* cat  = (bf16_t*)alloc((size_t)S * 512 * 2);
    bf16_t* y5   = (bf16_t*)alloc((size_t)S * 512 * 2);
    float*  stats = (float*)alloc((size_t)5 * 1024 * 4);   // [layer][sum|sumsq]
    float*  scbi  = (float*)alloc((size_t)5 * 1024 * 4);   // [layer][scale|shift]

    // --- init + weight conversion ---
    zero_f32_kernel<<<(5 * 1024 + 255) / 256, 256, 0, stream>>>(stats, 5 * 1024);
    cvt_f32_to_bf16_kernel<<<(64 * 64 + 255) / 256, 256, 0, stream>>>(Wf[1], Wb2, 64 * 64);
    cvt_f32_to_bf16_kernel<<<(128 * 64 + 255) / 256, 256, 0, stream>>>(Wf[2], Wb3, 128 * 64);
    cvt_f32_to_bf16_kernel<<<(256 * 128 + 255) / 256, 256, 0, stream>>>(Wf[3], Wb4, 256 * 128);
    cvt_f32_to_bf16_kernel<<<(512 * 512 + 1023) / 1024, 256, 0, stream>>>(Wf[4], Wb5, 512 * 512);

    // --- kNN graph ---
    knn_kernel<<<B * (N / 256), 256, 0, stream>>>(x, idx, B, N);

    // --- layer 1: edge features + W1 (VALU, Cin=6) ---
    edgeconv1_kernel<<<(int)(E / 128), 128, 0, stream>>>(x, idx, Wf[0], a1, B, N);

    auto bn_act = [&](bf16_t* y, int C, int layer, int catOff, long rowsCnt) {
        stats_kernel<<<256, 256, 0, stream>>>(y, rowsCnt * C, C, stats + layer * 1024);
        bnparam_kernel<<<(C + 63) / 64, 64, 0, stream>>>(
            stats + layer * 1024, gf[layer], bf[layer], (float)rowsCnt, C,
            scbi + layer * 1024);
        if (catOff >= 0)
            act_max_kernel<<<(int)(((long)S * C + 255) / 256), 256, 0, stream>>>(
                y, scbi + layer * 1024, cat, C, catOff, S);
    };

    bn_act(a1, 64, 0, 0, E);

    // --- layers 2-4: WMMA GEMMs over E edge-columns ---
    wmma_gemm_bf16_kernel<<<dim3((int)(E / 64), 64 / 64), 128, 64 * 64 * 2, stream>>>(
        Wb2, a1, a2, 64, 64, (int)E);
    bn_act(a2, 64, 1, 64, E);

    wmma_gemm_bf16_kernel<<<dim3((int)(E / 64), 128 / 64), 128, 64 * 64 * 2, stream>>>(
        Wb3, a2, a3, 64, 128, (int)E);
    bn_act(a3, 128, 2, 128, E);

    wmma_gemm_bf16_kernel<<<dim3((int)(E / 64), 256 / 64), 128, 64 * 128 * 2, stream>>>(
        Wb4, a3, a4, 128, 256, (int)E);
    bn_act(a4, 256, 3, 256, E);

    // --- layer 5: 512x512 WMMA GEMM over the S concat columns ---
    wmma_gemm_bf16_kernel<<<dim3(S / 64, 512 / 64), 128, 64 * 512 * 2, stream>>>(
        Wb5, cat, y5, 512, 512, S);
    stats_kernel<<<256, 256, 0, stream>>>(y5, (long)S * 512, 512, stats + 4 * 1024);
    bnparam_kernel<<<(512 + 63) / 64, 64, 0, stream>>>(
        stats + 4 * 1024, gf[4], bf[4], (float)S, 512, scbi + 4 * 1024);

    // --- BN+ReLU + transpose to [B][512][N] f32 ---
    final_kernel<<<dim3(N / 64, 512 / 64, B), 256, 0, stream>>>(
        y5, scbi + 4 * 1024, out, N);
}
